// Lighter_15461882265752
// MI455X (gfx1250) — compile-verified, matrix operand order checked by reading, not run on hardware
//
#include <hip/hip_runtime.h>
#include <hip/hip_bf16.h>

typedef __attribute__((ext_vector_type(16))) _Float16 v16h;
typedef __attribute__((ext_vector_type(8)))  _Float16 h8;
typedef __attribute__((ext_vector_type(4)))  _Float16 h4;
typedef __attribute__((ext_vector_type(8)))  float    v8f;
typedef __attribute__((ext_vector_type(4)))  unsigned int u32x4;
typedef __attribute__((ext_vector_type(8)))  int          i32x8;
typedef __attribute__((ext_vector_type(4)))  int          i32x4;

// ---------------------------------------------------------------------------
// WMMA helpers (CDNA5 16x16x32 f16 -> f32 accumulate)
// ---------------------------------------------------------------------------
__device__ inline v8f wmma32(v16h a, v16h b, v8f c) {
    return __builtin_amdgcn_wmma_f32_16x16x32_f16(
        false, a, false, b, (short)0, c, false, false);
}

// A fragment: 16x32 (MxK) f16 from row-major storage, leading dim `ld`.
__device__ inline v16h load_frag_a(const _Float16* p, int ld) {
    const int lane = threadIdx.x & 31;
    const int m  = lane & 15;
    const int kb = (lane >> 4) << 3;
    const h8 lo = *(const h8*)(p + m * ld + kb);
    const h8 hi = *(const h8*)(p + m * ld + kb + 16);
    v16h a;
#pragma unroll
    for (int i = 0; i < 8; ++i) { a[i] = lo[i]; a[8 + i] = hi[i]; }
    return a;
}

// B fragment from pre-swizzled storage (512 contiguous halves per fragment).
__device__ inline v16h load_frag_bsw(const _Float16* fragBase) {
    const int lane = threadIdx.x & 31;
    return *(const v16h*)(fragBase + lane * 16);
}

// C/D fragment -> f16 row-major store with optional ReLU.
__device__ inline void store_frag_f16(_Float16* p, int ld, v8f c, bool relu) {
    const int lane = threadIdx.x & 31;
    const int n  = lane & 15;
    const int mb = (lane >> 4) << 3;
#pragma unroll
    for (int v = 0; v < 8; ++v) {
        float x = c[v];
        if (relu) x = fmaxf(x, 0.0f);
        p[(mb + v) * ld + n] = (_Float16)x;
    }
}

// ---------------------------------------------------------------------------
// TDM: issue a 2D tensor tile load (rows x 32 cols of f16) into LDS.
// D# group layout per CDNA5 ISA ch.8: group0 = {flags, lds_addr, global_addr,
// type}, group1 = {data_size, tensor dims, tile dims, strides}. cols beyond
// `colsValid` are OOB -> hardware zero-fills (used for the K=900 tail).
// This toolchain exposes the 6-arg builtin (g0, g1, g2, g3, g4, cpol).
// ---------------------------------------------------------------------------
__device__ inline void tdm_load_tile_f16(const _Float16* gsrc, unsigned ldsOff,
                                         unsigned colsValid, unsigned rows,
                                         unsigned ldElems) {
    const unsigned long long ga = (unsigned long long)(size_t)gsrc;
    u32x4 g0;
    g0[0] = 1u;                                       // count=1, user D#
    g0[1] = ldsOff;                                   // lds_addr (bytes)
    g0[2] = (unsigned)(ga & 0xffffffffull);           // global_addr[31:0]
    g0[3] = (unsigned)((ga >> 32) & 0x1ffffffull)     // global_addr[56:32]
            | (2u << 30);                             // type = 2 (image)
    i32x8 g1;
    g1[0] = (int)(1u << 16);                          // data_size = 1 -> 2B
    g1[1] = (int)((colsValid & 0xffffu) << 16);       // tensor_dim0[15:0]
    g1[2] = (int)((colsValid >> 16) | ((rows & 0xffffu) << 16)); // dim0 hi | dim1 lo
    g1[3] = (int)((rows >> 16) | (32u << 16));        // dim1 hi | tile_dim0 = 32
    g1[4] = (int)rows;                                // tile_dim1 = rows, tile_dim2 = 0
    g1[5] = (int)ldElems;                             // tensor_dim0_stride[31:0]
    g1[6] = 0;                                        // stride hi, dim1_stride lo
    g1[7] = 0;
    const i32x4 z4 = {0, 0, 0, 0};
    const i32x8 z8 = {0, 0, 0, 0, 0, 0, 0, 0};
    __builtin_amdgcn_tensor_load_to_lds(g0, g1, z4, z4, z8, 0);
}

// ---------------------------------------------------------------------------
// Weight convert + swizzle: W fp32 [N(out) x K(in)] -> f16 B fragments.
// ---------------------------------------------------------------------------
__global__ __launch_bounds__(256) void convert_w_swz_kernel(
    const float* __restrict__ W, _Float16* __restrict__ B,
    int N, int K, int Kpad) {
    const int i = blockIdx.x * 256 + threadIdx.x;
    if (i >= Kpad * N) return;
    const int frag = i >> 9;
    const int r    = i & 511;
    const int lane = r >> 4;
    const int e    = r & 15;
    const int nl = lane & 15;
    const int kb = (lane >> 4) << 3;
    const int pi = e >> 1, odd = e & 1;
    const int kl = kb + ((pi < 4) ? 2 * pi : 16 + 2 * (pi - 4)) + odd;
    const int n16 = N >> 4;
    const int kt = frag / n16, ntile = frag - kt * n16;
    const int k = kt * 32 + kl, n = ntile * 16 + nl;
    B[i] = (k < K) ? (_Float16)W[(long)n * K + k] : (_Float16)0.0f;
}

// ---------------------------------------------------------------------------
// Fused block MLP: rows = batch*100; 9 -> 128 -> 32 -> 16 -> 1
// ---------------------------------------------------------------------------
__global__ __launch_bounds__(256) void block_mlp_kernel(
    const float* __restrict__ X,
    const _Float16* __restrict__ gw1, const _Float16* __restrict__ gw2,
    const _Float16* __restrict__ gw3, const float* __restrict__ w4,
    float* __restrict__ Bv) {
    __shared__ __align__(32) _Float16 w1s[32 * 128];
    __shared__ __align__(32) _Float16 w2s[128 * 32];
    __shared__ __align__(32) _Float16 w3s[32 * 16];
    __shared__ __align__(32) _Float16 actA[8][16 * 128];
    __shared__ __align__(32) _Float16 actB[8][16 * 32];

    const int tid = threadIdx.x;
    for (int i = tid * 8; i < 32 * 128; i += 256 * 8) *(h8*)(w1s + i) = *(const h8*)(gw1 + i);
    for (int i = tid * 8; i < 128 * 32; i += 256 * 8) *(h8*)(w2s + i) = *(const h8*)(gw2 + i);
    for (int i = tid * 8; i < 32 * 16;  i += 256 * 8) *(h8*)(w3s + i) = *(const h8*)(gw3 + i);
    __syncthreads();

    const int wave = tid >> 5;
    const int lane = tid & 31;
    _Float16* reg  = actA[wave];
    _Float16* regB = actB[wave];
    const int rowBase = (blockIdx.x * 8 + wave) * 16;

    for (int i = lane; i < 16 * 32; i += 32) reg[i] = (_Float16)0.0f;
    for (int idx = lane; idx < 16 * 9; idx += 32) {
        const int r = idx / 9, j = idx - r * 9;
        const int grow = rowBase + r;
        const int b = grow / 100, node = grow - b * 100;
        const int nr = node / 10, nc = node - nr * 10;
        const int pi = j / 3, pj = j - pi * 3;
        const int pix = (nr * 3 + pi) * 30 + (nc * 3 + pj);
        const float v = X[(long)b * 900 + pix] * 255.0f + 100.0f;
        reg[r * 32 + j] = (_Float16)v;
    }
    v16h a0 = load_frag_a(reg, 32);

#pragma unroll
    for (int nt = 0; nt < 8; ++nt) {           // L1: K=32, N=128
        v16h b = load_frag_bsw(w1s + nt * 512);
        v8f c = {};
        c = wmma32(a0, b, c);
        store_frag_f16(reg + nt * 16, 128, c, true);
    }
#pragma unroll
    for (int nt = 0; nt < 2; ++nt) {           // L2: K=128, N=32
        v8f c = {};
#pragma unroll
        for (int ks = 0; ks < 4; ++ks) {
            v16h a = load_frag_a(reg + ks * 32, 128);
            v16h b = load_frag_bsw(w2s + (ks * 2 + nt) * 512);
            c = wmma32(a, b, c);
        }
        store_frag_f16(regB + nt * 16, 32, c, true);
    }
    {                                           // L3: K=32, N=16
        v16h a = load_frag_a(regB, 32);
        v16h b = load_frag_bsw(w3s);
        v8f c = {};
        c = wmma32(a, b, c);
        store_frag_f16(reg, 16, c, true);
    }
    if (lane < 16) {                            // L4: 16 -> 1
        float s = 0.0f;
#pragma unroll
        for (int kk = 0; kk < 16; ++kk)
            s += (float)reg[lane * 16 + kk] * w4[kk];
        Bv[rowBase + lane] = s;
    }
}

// ---------------------------------------------------------------------------
// Per-row min/max normalize of Bv.
// ---------------------------------------------------------------------------
__global__ __launch_bounds__(128) void norm_bv_kernel(
    const float* __restrict__ Bv, float* __restrict__ BvN) {
    __shared__ float smin[128], smax[128];
    const int row = blockIdx.x, t = threadIdx.x;
    const float v = (t < 100) ? Bv[(long)row * 100 + t] : 0.0f;
    smin[t] = (t < 100) ? v : 3.4e38f;
    smax[t] = (t < 100) ? v : -3.4e38f;
    __syncthreads();
    for (int s = 64; s > 0; s >>= 1) {
        if (t < s) {
            smin[t] = fminf(smin[t], smin[t + s]);
            smax[t] = fmaxf(smax[t], smax[t + s]);
        }
        __syncthreads();
    }
    const float mn = smin[0], mx = smax[0];
    if (t < 100)
        BvN[(long)row * 100 + t] = (v - mn) / (mx - mn) * 255.0f + 100.0f;
}

// ---------------------------------------------------------------------------
// Fused attn MLP: 25 -> 128 -> 64 -> 32 -> 16 -> 1
// ---------------------------------------------------------------------------
__global__ __launch_bounds__(128) void attn_mlp_kernel(
    const float* __restrict__ BvN,
    const _Float16* __restrict__ gw1, const _Float16* __restrict__ gw2,
    const _Float16* __restrict__ gw3, const _Float16* __restrict__ gw4,
    const float* __restrict__ w5, float* __restrict__ E) {
    __shared__ __align__(32) _Float16 w1s[32 * 128];
    __shared__ __align__(32) _Float16 w2s[128 * 64];
    __shared__ __align__(32) _Float16 w3s[64 * 32];
    __shared__ __align__(32) _Float16 w4s[32 * 16];
    __shared__ __align__(32) _Float16 actA[4][16 * 128];
    __shared__ __align__(32) _Float16 actB[4][16 * 64];

    const int tid = threadIdx.x;
    for (int i = tid * 8; i < 32 * 128; i += 128 * 8) *(h8*)(w1s + i) = *(const h8*)(gw1 + i);
    for (int i = tid * 8; i < 128 * 64; i += 128 * 8) *(h8*)(w2s + i) = *(const h8*)(gw2 + i);
    for (int i = tid * 8; i < 64 * 32;  i += 128 * 8) *(h8*)(w3s + i) = *(const h8*)(gw3 + i);
    for (int i = tid * 8; i < 32 * 16;  i += 128 * 8) *(h8*)(w4s + i) = *(const h8*)(gw4 + i);
    __syncthreads();

    const int wave = tid >> 5;
    const int lane = tid & 31;
    _Float16* regA = actA[wave];
    _Float16* regB = actB[wave];
    const int rowBase = (blockIdx.x * 4 + wave) * 16;

    for (int i = lane; i < 16 * 32; i += 32) regB[i] = (_Float16)0.0f;
    for (int idx = lane; idx < 16 * 25; idx += 32) {
        const int r = idx / 25, cell = idx - r * 25;
        const int grow = rowBase + r;
        const int b = grow / 100, node = grow - b * 100;
        const int nr = node / 10, nc = node - nr * 10;
        const int di = cell / 5 - 2, dj = cell - (cell / 5) * 5 - 2;
        const int rr = nr + di, cc = nc + dj;
        float v = -1.0f;
        if (rr >= 0 && rr < 10 && cc >= 0 && cc < 10)
            v = BvN[(long)b * 100 + rr * 10 + cc];
        regB[r * 32 + cell] = (_Float16)v;
    }
    v16h a0 = load_frag_a(regB, 32);

#pragma unroll
    for (int nt = 0; nt < 8; ++nt) {           // L1: K=32, N=128
        v16h b = load_frag_bsw(w1s + nt * 512);
        v8f c = {};
        c = wmma32(a0, b, c);
        store_frag_f16(regA + nt * 16, 128, c, true);
    }
#pragma unroll
    for (int nt = 0; nt < 4; ++nt) {           // L2: K=128, N=64
        v8f c = {};
#pragma unroll
        for (int ks = 0; ks < 4; ++ks) {
            v16h a = load_frag_a(regA + ks * 32, 128);
            v16h b = load_frag_bsw(w2s + (ks * 4 + nt) * 512);
            c = wmma32(a, b, c);
        }
        store_frag_f16(regB + nt * 16, 64, c, true);
    }
#pragma unroll
    for (int nt = 0; nt < 2; ++nt) {           // L3: K=64, N=32
        v8f c = {};
#pragma unroll
        for (int ks = 0; ks < 2; ++ks) {
            v16h a = load_frag_a(regB + ks * 32, 64);
            v16h b = load_frag_bsw(w3s + (ks * 2 + nt) * 512);
            c = wmma32(a, b, c);
        }
        store_frag_f16(regA + nt * 16, 32, c, true);
    }
    {                                           // L4: K=32, N=16
        v16h a = load_frag_a(regA, 32);
        v16h b = load_frag_bsw(w4s);
        v8f c = {};
        c = wmma32(a, b, c);
        store_frag_f16(regB, 16, c, true);
    }
    if (lane < 16) {                            // L5: 16 -> 1
        float s = 0.0f;
#pragma unroll
        for (int kk = 0; kk < 16; ++kk)
            s += (float)regB[lane * 16 + kk] * w5[kk];
        E[rowBase + lane] = s;
    }
}

// ---------------------------------------------------------------------------
// Normalize E, ReLU(E - bias^2), 3x3 upsample, scale (X*255+100) -> f16.
// ---------------------------------------------------------------------------
__global__ __launch_bounds__(128) void scale_kernel(
    const float* __restrict__ X, const float* __restrict__ E,
    const float* __restrict__ relu_bias, _Float16* __restrict__ XsE) {
    __shared__ float smin[128], smax[128], se[100];
    const int row = blockIdx.x, t = threadIdx.x;
    const float v = (t < 100) ? E[(long)row * 100 + t] : 0.0f;
    smin[t] = (t < 100) ? v : 3.4e38f;
    smax[t] = (t < 100) ? v : -3.4e38f;
    __syncthreads();
    for (int s = 64; s > 0; s >>= 1) {
        if (t < s) {
            smin[t] = fminf(smin[t], smin[t + s]);
            smax[t] = fmaxf(smax[t], smax[t + s]);
        }
        __syncthreads();
    }
    const float mn = smin[0], mx = smax[0];
    const float thr = relu_bias[0] * relu_bias[0];
    if (t < 100) se[t] = fmaxf((v - mn) / (mx - mn) - thr, 0.0f);
    __syncthreads();
    for (int p = t; p < 900; p += 128) {
        const int pr = p / 30, pc = p - pr * 30;
        const int node = (pr / 3) * 10 + (pc / 3);
        const float xs = X[(long)row * 900 + p] * 255.0f + 100.0f;
        XsE[(long)row * 900 + p] = (_Float16)(xs * se[node]);
    }
}

// ---------------------------------------------------------------------------
// Tiled WMMA GEMM with TDM-staged A tiles (double-buffered).
// A row-major f16 [MxK]; B pre-swizzled fragments; wave tile 16 x (16*NT).
// Wave 0 drives the Tensor Data Mover; TENSORcnt + barrier synchronize.
// ---------------------------------------------------------------------------
template <int NT>
__global__ __launch_bounds__(256) void gemm_f16_kernel(
    const _Float16* __restrict__ A, const _Float16* __restrict__ Bsw,
    _Float16* __restrict__ C, int M, int K, int N, int relu) {
    __shared__ __align__(32) _Float16 As[2][64 * 32];
    const int tid  = threadIdx.x;
    const int wave = tid >> 5, lane = tid & 31;
    const int wm = wave >> 1, wn = wave & 1;
    const long m0 = (long)blockIdx.x * 64;
    const int  n0 = blockIdx.y * (32 * NT);

    v8f c[NT];
#pragma unroll
    for (int i = 0; i < NT; ++i) c[i] = (v8f){};

    const int ksteps = (K + 31) / 32;
    const int nfrag  = N >> 4;
    const bool issuer = (__builtin_amdgcn_readfirstlane(wave) == 0);

    if (issuer)   // prime the pipeline: tile 0 -> buffer 0
        tdm_load_tile_f16(A + m0 * K, (unsigned)(size_t)&As[0][0],
                          (unsigned)K, 64u, (unsigned)K);

    for (int kt = 0; kt < ksteps; ++kt) {
        const int cur = kt & 1;
        if (issuer) {
            if (kt + 1 < ksteps) {
                tdm_load_tile_f16(A + m0 * K + (kt + 1) * 32,
                                  (unsigned)(size_t)&As[cur ^ 1][0],
                                  (unsigned)(K - (kt + 1) * 32), 64u,
                                  (unsigned)K);
                __builtin_amdgcn_s_wait_tensorcnt(1);  // tile kt resident
            } else {
                __builtin_amdgcn_s_wait_tensorcnt(0);
            }
        }
        __syncthreads();
        v16h a = load_frag_a(&As[cur][0] + wm * 16 * 32, 32);
        const _Float16* bbase =
            Bsw + ((long)kt * nfrag + (n0 >> 4) + wn * NT) * 512 + lane * 16;
#pragma unroll
        for (int nt = 0; nt < NT; ++nt) {
            v16h b = *(const v16h*)(bbase + nt * 512);
            c[nt] = wmma32(a, b, c[nt]);
        }
        __syncthreads();
    }
    const int ncol = lane & 15;
    const int mb   = (lane >> 4) << 3;
#pragma unroll
    for (int nt = 0; nt < NT; ++nt) {
#pragma unroll
        for (int v = 0; v < 8; ++v) {
            float x = c[nt][v];
            if (relu) x = fmaxf(x, 0.0f);
            const long r = m0 + wm * 16 + mb + v;
            C[r * N + n0 + (wn * NT + nt) * 16 + ncol] = (_Float16)x;
        }
    }
}

// ---------------------------------------------------------------------------
// Final classifier layer 64 -> 10, fp32 out.
// ---------------------------------------------------------------------------
__global__ __launch_bounds__(256) void cls_last_kernel(
    const _Float16* __restrict__ act, const float* __restrict__ W5,
    float* __restrict__ out) {
    const int gt = blockIdx.x * 256 + threadIdx.x;
    if (gt >= 16384 * 10) return;
    const int row = gt / 10, o = gt - row * 10;
    float s = 0.0f;
#pragma unroll
    for (int k = 0; k < 64; ++k)
        s += (float)act[(long)row * 64 + k] * W5[o * 64 + k];
    out[gt] = s;
}

// ---------------------------------------------------------------------------
// Workspace layout (bytes, 256-aligned)
// ---------------------------------------------------------------------------
static constexpr size_t al256(size_t x) { return (x + 255) & ~(size_t)255; }
static constexpr size_t OFF_WB1 = 0;
static constexpr size_t OFF_WB2 = al256(OFF_WB1 + (size_t)32 * 128 * 2);
static constexpr size_t OFF_WB3 = al256(OFF_WB2 + (size_t)128 * 32 * 2);
static constexpr size_t OFF_WA1 = al256(OFF_WB3 + (size_t)32 * 16 * 2);
static constexpr size_t OFF_WA2 = al256(OFF_WA1 + (size_t)32 * 128 * 2);
static constexpr size_t OFF_WA3 = al256(OFF_WA2 + (size_t)128 * 64 * 2);
static constexpr size_t OFF_WA4 = al256(OFF_WA3 + (size_t)64 * 32 * 2);
static constexpr size_t OFF_WC1 = al256(OFF_WA4 + (size_t)32 * 16 * 2);
static constexpr size_t OFF_WC2 = al256(OFF_WC1 + (size_t)928 * 1024 * 2);
static constexpr size_t OFF_WC3 = al256(OFF_WC2 + (size_t)1024 * 512 * 2);
static constexpr size_t OFF_WC4 = al256(OFF_WC3 + (size_t)512 * 128 * 2);
static constexpr size_t OFF_BV  = al256(OFF_WC4 + (size_t)128 * 64 * 2);
static constexpr size_t OFF_BVN = al256(OFF_BV  + (size_t)16384 * 100 * 4);
static constexpr size_t OFF_E   = al256(OFF_BVN + (size_t)16384 * 100 * 4);
static constexpr size_t OFF_XSE = al256(OFF_E   + (size_t)16384 * 100 * 4);
static constexpr size_t OFF_A1  = al256(OFF_XSE + (size_t)16384 * 900 * 2);
static constexpr size_t OFF_A2  = al256(OFF_A1  + (size_t)16384 * 1024 * 2);
static constexpr size_t OFF_A3  = al256(OFF_A2  + (size_t)16384 * 512 * 2);
static constexpr size_t OFF_A4  = al256(OFF_A3  + (size_t)16384 * 128 * 2);

extern "C" void kernel_launch(void* const* d_in, const int* in_sizes, int n_in,
                              void* d_out, int out_size, void* d_ws, size_t ws_size,
                              hipStream_t stream) {
    const float* X   = (const float*)d_in[0];
    const float* bw0 = (const float*)d_in[2];
    const float* bw1 = (const float*)d_in[3];
    const float* bw2 = (const float*)d_in[4];
    const float* bw3 = (const float*)d_in[5];
    const float* aw0 = (const float*)d_in[6];
    const float* aw1 = (const float*)d_in[7];
    const float* aw2 = (const float*)d_in[8];
    const float* aw3 = (const float*)d_in[9];
    const float* aw4 = (const float*)d_in[10];
    const float* cw0 = (const float*)d_in[11];
    const float* cw1 = (const float*)d_in[12];
    const float* cw2 = (const float*)d_in[13];
    const float* cw3 = (const float*)d_in[14];
    const float* cw4 = (const float*)d_in[15];
    const float* rb  = (const float*)d_in[16];

    char* ws = (char*)d_ws;
    _Float16* WB1 = (_Float16*)(ws + OFF_WB1);
    _Float16* WB2 = (_Float16*)(ws + OFF_WB2);
    _Float16* WB3 = (_Float16*)(ws + OFF_WB3);
    _Float16* WA1 = (_Float16*)(ws + OFF_WA1);
    _Float16* WA2 = (_Float16*)(ws + OFF_WA2);
    _Float16* WA3 = (_Float16*)(ws + OFF_WA3);
    _Float16* WA4 = (_Float16*)(ws + OFF_WA4);
    _Float16* WC1 = (_Float16*)(ws + OFF_WC1);
    _Float16* WC2 = (_Float16*)(ws + OFF_WC2);
    _Float16* WC3 = (_Float16*)(ws + OFF_WC3);
    _Float16* WC4 = (_Float16*)(ws + OFF_WC4);
    float*    Bv  = (float*)(ws + OFF_BV);
    float*    BvN = (float*)(ws + OFF_BVN);
    float*    E   = (float*)(ws + OFF_E);
    _Float16* XsE = (_Float16*)(ws + OFF_XSE);
    _Float16* A1  = (_Float16*)(ws + OFF_A1);
    _Float16* A2  = (_Float16*)(ws + OFF_A2);
    _Float16* A3  = (_Float16*)(ws + OFF_A3);
    _Float16* A4  = (_Float16*)(ws + OFF_A4);

    auto cvt = [&](const float* W, _Float16* B, int N, int K, int Kpad) {
        const int total = Kpad * N;
        convert_w_swz_kernel<<<(total + 255) / 256, 256, 0, stream>>>(W, B, N, K, Kpad);
    };
    cvt(bw0, WB1, 128, 9, 32);
    cvt(bw1, WB2, 32, 128, 128);
    cvt(bw2, WB3, 16, 32, 32);
    cvt(aw0, WA1, 128, 25, 32);
    cvt(aw1, WA2, 64, 128, 128);
    cvt(aw2, WA3, 32, 64, 64);
    cvt(aw3, WA4, 16, 32, 32);
    cvt(cw0, WC1, 1024, 900, 928);
    cvt(cw1, WC2, 512, 1024, 1024);
    cvt(cw2, WC3, 128, 512, 512);
    cvt(cw3, WC4, 64, 128, 128);

    block_mlp_kernel<<<12800, 256, 0, stream>>>(X, WB1, WB2, WB3, bw3, Bv);
    norm_bv_kernel<<<16384, 128, 0, stream>>>(Bv, BvN);
    attn_mlp_kernel<<<25600, 128, 0, stream>>>(BvN, WA1, WA2, WA3, WA4, aw4, E);
    scale_kernel<<<16384, 128, 0, stream>>>(X, E, rb, XsE);

    gemm_f16_kernel<4><<<dim3(256, 8), 256, 0, stream>>>(XsE, WC1, A1, 16384, 900,  1024, 1);
    gemm_f16_kernel<4><<<dim3(256, 4), 256, 0, stream>>>(A1,  WC2, A2, 16384, 1024, 512,  1);
    gemm_f16_kernel<4><<<dim3(256, 1), 256, 0, stream>>>(A2,  WC3, A3, 16384, 512,  128,  1);
    gemm_f16_kernel<2><<<dim3(256, 1), 256, 0, stream>>>(A3,  WC4, A4, 16384, 128,  64,   1);
    cls_last_kernel<<<640, 256, 0, stream>>>(A4, cw4, (float*)d_out);
}
